// CausalSelfAttention_67327907332415
// MI455X (gfx1250) — matrix operand order repaired and checked by reading, not compiled, not run on hardware
//
#include <hip/hip_runtime.h>
#include <hip/hip_bf16.h>

// Problem constants (reference: B=4, T=2048, D=1024, H=16, DK=64)
#define BB  4
#define TT  2048
#define DD  1024
#define HH  16
#define DKK 64

typedef __attribute__((ext_vector_type(16))) __bf16 v16bf;
typedef __attribute__((ext_vector_type(8)))  __bf16 v8bf;
typedef __attribute__((ext_vector_type(8)))  float  v8f;

// round-to-nearest-even fp32 -> bf16 without relying on __bf16 conversion support
static __device__ __forceinline__ __bf16 f2bf(float f) {
  unsigned u = __builtin_bit_cast(unsigned, f);
  u += 0x7FFFu + ((u >> 16) & 1u);
  unsigned short s = (unsigned short)(u >> 16);
  return __builtin_bit_cast(__bf16, s);
}

static __device__ __forceinline__ v16bf cat8(v8bf lo, v8bf hi) {
  return __builtin_shufflevector(lo, hi, 0,1,2,3,4,5,6,7,8,9,10,11,12,13,14,15);
}

static __device__ __forceinline__ v8f wmma_bf16(v16bf a, v16bf b, v8f c) {
  // D(f32 16x16) = A(bf16 16x32) * B(bf16 32x16) + C
  return __builtin_amdgcn_wmma_f32_16x16x32_bf16(false, a, false, b, (short)0, c,
                                                 false, false);
}

// ---------------------------------------------------------------------------
// 1) elementwise fp32 -> bf16
__global__ void k_f32_to_bf16(const float* __restrict__ src,
                              __bf16* __restrict__ dst, int n) {
  int i = blockIdx.x * blockDim.x + threadIdx.x;
  if (i < n) dst[i] = f2bf(src[i]);
}

// 2) src [K][N] fp32 -> dst [N][K] bf16 (weight transpose so WMMA B-fragments
//    are K-contiguous per lane)
__global__ void k_transpose_to_bf16(const float* __restrict__ src,
                                    __bf16* __restrict__ dst, int K, int N) {
  int i = blockIdx.x * blockDim.x + threadIdx.x;
  if (i < K * N) {
    int n = i / K, k = i - n * K;
    dst[i] = f2bf(src[(size_t)k * N + n]);
  }
}

// ---------------------------------------------------------------------------
// 3) QKV GEMM: [M=B*T, K=D] x [N=3D, K=D]^T  + bias, scatter epilogue into
//    q [B,H,T,64], k [B,H,T,64], vT [B,H,64,T]  (all bf16)
__global__ __launch_bounds__(128) void k_gemm_qkv(
    const __bf16* __restrict__ A, const __bf16* __restrict__ Wt,
    const float* __restrict__ bias,
    __bf16* __restrict__ qb, __bf16* __restrict__ kbuf, __bf16* __restrict__ vtb) {
  const int lane = threadIdx.x & 31;
  const int wave = threadIdx.x >> 5;
  const int hf   = lane >> 4;          // half-wave: 0 or 1
  const int l16  = lane & 15;
  const int m0 = blockIdx.y * 64 + (wave >> 1) * 32;
  const int n0 = blockIdx.x * 64 + (wave & 1) * 32;
  const int abase = hf ? 16 : 0;       // A frag K-chunk base per half-wave
  const int bbase = hf ? 16 : 0;       // B frag K base per half-wave

  v8f c[2][2] = {};
  for (int kk = 0; kk < DD; kk += 32) {
    v16bf a[2], bfr[2];
#pragma unroll
    for (int i = 0; i < 2; ++i) {
      const __bf16* ap = A + (size_t)(m0 + i * 16 + l16) * DD + kk + abase;
      a[i] = cat8(*(const v8bf*)ap, *(const v8bf*)(ap + 8));
    }
#pragma unroll
    for (int j = 0; j < 2; ++j) {
      const __bf16* bp = Wt + (size_t)(n0 + j * 16 + l16) * DD + kk + bbase;
      __builtin_prefetch(bp + 32, 0, 0);   // global_prefetch_b8 next k-step
      bfr[j] = *(const v16bf*)bp;
    }
#pragma unroll
    for (int i = 0; i < 2; ++i)
#pragma unroll
      for (int j = 0; j < 2; ++j)
        c[i][j] = wmma_bf16(a[i], bfr[j], c[i][j]);
  }
#pragma unroll
  for (int j = 0; j < 2; ++j) {
    const int n  = n0 + j * 16 + l16;
    const float bv = bias[n];
    const int sel = n >> 10;            // 0=q 1=k 2=v
    const int nn  = n & 1023;
    const int h = nn >> 6, dk = nn & 63;
#pragma unroll
    for (int i = 0; i < 2; ++i)
#pragma unroll
      for (int e = 0; e < 8; ++e) {
        const int m = m0 + i * 16 + e + hf * 8;     // C layout row
        const int bI = m >> 11, t = m & 2047;
        const __bf16 val = f2bf(c[i][j][e] + bv);
        const size_t bh = (size_t)bI * HH + h;
        if (sel == 0)      qb  [(bh * TT  + t ) * DKK + dk] = val;
        else if (sel == 1) kbuf[(bh * TT  + t ) * DKK + dk] = val;
        else               vtb [(bh * DKK + dk) * TT  + t ] = val;
      }
  }
}

// ---------------------------------------------------------------------------
// 4) causal flash attention: 1 wave = 16 query rows, key blocks of 64,
//    online softmax in fp32, P staged through LDS (C-layout -> A-layout),
//    output written bf16 into ctx [B,T,D] with heads merged.
__global__ __launch_bounds__(128) void k_attn(
    const __bf16* __restrict__ qb, const __bf16* __restrict__ kbuf,
    const __bf16* __restrict__ vtb, __bf16* __restrict__ ctx) {
  __shared__ __bf16 lds[4 * 16 * 72];   // per-wave 16 x 64 P tile, stride 72
  const int lane = threadIdx.x & 31;
  const int wave = threadIdx.x >> 5;
  const int hf   = lane >> 4;
  const int l16  = lane & 15;
  const int b = blockIdx.z, h = blockIdx.y;
  const int q0 = blockIdx.x * 64 + wave * 16;   // this wave's query rows

  const size_t bh = (size_t)b * HH + h;
  const __bf16* Q = qb   + bh * TT  * DKK;
  const __bf16* K = kbuf + bh * TT  * DKK;
  const __bf16* V = vtb  + bh * DKK * TT;
  __bf16* P = lds + wave * (16 * 72);

  // Q A-fragments for dk-steps {0..31},{32..63}
  v16bf qa[2];
  {
    const __bf16* qp = Q + (size_t)(q0 + l16) * DKK + (hf ? 16 : 0);
#pragma unroll
    for (int s = 0; s < 2; ++s)
      qa[s] = cat8(*(const v8bf*)(qp + s * 32), *(const v8bf*)(qp + s * 32 + 8));
  }

  v8f O[4] = {};
  float mrow[8], lrow[8];
#pragma unroll
  for (int e = 0; e < 8; ++e) { mrow[e] = -1e30f; lrow[e] = 0.0f; }

  for (int kb = 0; kb <= q0 + 15; kb += 64) {
    // scores S = (Q K^T): 4 fragments of 16 keys, K-dim = 64 (2 wmma each)
    v8f S[4];
#pragma unroll
    for (int j = 0; j < 4; ++j) {
      const int key = kb + j * 16 + l16;
      const __bf16* kp = K + (size_t)key * DKK + (hf ? 16 : 0);
      v8f acc = {};
      acc = wmma_bf16(qa[0], *(const v16bf*)kp, acc);
      acc = wmma_bf16(qa[1], *(const v16bf*)(kp + 32), acc);
      S[j] = acc;
    }
    // scale 1/sqrt(64) + causal mask
#pragma unroll
    for (int j = 0; j < 4; ++j) {
      const int key = kb + j * 16 + l16;
#pragma unroll
      for (int e = 0; e < 8; ++e) {
        const int row = q0 + e + hf * 8;
        const float s = S[j][e] * 0.125f;
        S[j][e] = (key <= row) ? s : -1e30f;
      }
    }
    // online softmax update (per-row stats replicated across 16-lane half)
#pragma unroll
    for (int e = 0; e < 8; ++e) {
      float mx = fmaxf(fmaxf(S[0][e], S[1][e]), fmaxf(S[2][e], S[3][e]));
#pragma unroll
      for (int d = 1; d < 16; d <<= 1) mx = fmaxf(mx, __shfl_xor(mx, d, 32));
      const float mnew  = fmaxf(mrow[e], mx);
      const float alpha = __expf(mrow[e] - mnew);
      float psum = 0.0f;
#pragma unroll
      for (int j = 0; j < 4; ++j) {
        const float p = __expf(S[j][e] - mnew);
        S[j][e] = p;
        psum += p;
      }
#pragma unroll
      for (int d = 1; d < 16; d <<= 1) psum += __shfl_xor(psum, d, 32);
      lrow[e] = lrow[e] * alpha + psum;
      mrow[e] = mnew;
#pragma unroll
      for (int jj = 0; jj < 4; ++jj) O[jj][e] *= alpha;
    }
    // P (C/D layout) -> LDS (row-major 16x64, stride 72) as bf16
#pragma unroll
    for (int j = 0; j < 4; ++j)
#pragma unroll
      for (int e = 0; e < 8; ++e)
        P[(e + hf * 8) * 72 + j * 16 + l16] = f2bf(S[j][e]);

    asm volatile("s_wait_dscnt 0" ::: "memory");   // LDS store->load ordering

    // O += P @ V : K-dim = 64 keys (2 steps of 32), 4 dk-column tiles
#pragma unroll
    for (int s = 0; s < 2; ++s) {
      const __bf16* pp = P + l16 * 72 + s * 32 + (hf ? 16 : 0);
      const v16bf pa = cat8(*(const v8bf*)pp, *(const v8bf*)(pp + 8));
#pragma unroll
      for (int jj = 0; jj < 4; ++jj) {
        const int dk = jj * 16 + l16;
        const __bf16* vp = V + (size_t)dk * TT + kb + s * 32 + (hf ? 16 : 0);
        O[jj] = wmma_bf16(pa, *(const v16bf*)vp, O[jj]);
      }
    }
  }
  // normalize and emit ctx [B,T,D] (heads merged)
#pragma unroll
  for (int e = 0; e < 8; ++e) {
    const float inv = 1.0f / lrow[e];
    const int row = q0 + e + hf * 8;
#pragma unroll
    for (int jj = 0; jj < 4; ++jj) {
      const int col = h * DKK + jj * 16 + l16;
      ctx[((size_t)b * TT + row) * DD + col] = f2bf(O[jj][e] * inv);
    }
  }
}

// ---------------------------------------------------------------------------
// 5) output GEMM: ctx[M,1024] x Wout^T[1024,1024] + b_out -> fp32 d_out
__global__ __launch_bounds__(128) void k_gemm_out(
    const __bf16* __restrict__ A, const __bf16* __restrict__ Wt,
    const float* __restrict__ bias, float* __restrict__ out) {
  const int lane = threadIdx.x & 31;
  const int wave = threadIdx.x >> 5;
  const int hf = lane >> 4, l16 = lane & 15;
  const int m0 = blockIdx.y * 64 + (wave >> 1) * 32;
  const int n0 = blockIdx.x * 64 + (wave & 1) * 32;
  const int abase = hf ? 16 : 0, bbase = hf ? 16 : 0;

  v8f c[2][2] = {};
  for (int kk = 0; kk < DD; kk += 32) {
    v16bf a[2], bfr[2];
#pragma unroll
    for (int i = 0; i < 2; ++i) {
      const __bf16* ap = A + (size_t)(m0 + i * 16 + l16) * DD + kk + abase;
      a[i] = cat8(*(const v8bf*)ap, *(const v8bf*)(ap + 8));
    }
#pragma unroll
    for (int j = 0; j < 2; ++j) {
      const __bf16* bp = Wt + (size_t)(n0 + j * 16 + l16) * DD + kk + bbase;
      __builtin_prefetch(bp + 32, 0, 0);
      bfr[j] = *(const v16bf*)bp;
    }
#pragma unroll
    for (int i = 0; i < 2; ++i)
#pragma unroll
      for (int j = 0; j < 2; ++j)
        c[i][j] = wmma_bf16(a[i], bfr[j], c[i][j]);
  }
#pragma unroll
  for (int j = 0; j < 2; ++j) {
    const int n = n0 + j * 16 + l16;
    const float bv = bias[n];
#pragma unroll
    for (int i = 0; i < 2; ++i)
#pragma unroll
      for (int e = 0; e < 8; ++e) {
        const int m = m0 + i * 16 + e + hf * 8;
        out[(size_t)m * DD + n] = c[i][j][e] + bv;
      }
  }
}

// ---------------------------------------------------------------------------
extern "C" void kernel_launch(void* const* d_in, const int* in_sizes, int n_in,
                              void* d_out, int out_size, void* d_ws, size_t ws_size,
                              hipStream_t stream) {
  const float* x    = (const float*)d_in[0];
  const float* Wqkv = (const float*)d_in[1];
  const float* bqkv = (const float*)d_in[2];
  const float* Wout = (const float*)d_in[3];
  const float* bout = (const float*)d_in[4];
  float* out = (float*)d_out;

  // workspace layout (bytes), ~88 MB total
  char* ws = (char*)d_ws;
  __bf16* xb    = (__bf16*)(ws +        0);  // B*T*D bf16        (16 MB)
  __bf16* wqkvT = (__bf16*)(ws + 16777216);  // [3D][D] bf16      ( 6 MB)
  __bf16* woutT = (__bf16*)(ws + 23068672);  // [D][D] bf16       ( 2 MB)
  __bf16* qb    = (__bf16*)(ws + 25165824);  // [B,H,T,64] bf16   (16 MB)
  __bf16* kbuf  = (__bf16*)(ws + 41943040);  // [B,H,T,64] bf16   (16 MB)
  __bf16* vtb   = (__bf16*)(ws + 58720256);  // [B,H,64,T] bf16   (16 MB)
  __bf16* ctx   = (__bf16*)(ws + 75497472);  // [B,T,D] bf16      (16 MB)

  {
    const int n = BB * TT * DD;
    k_f32_to_bf16<<<n / 256, 256, 0, stream>>>(x, xb, n);
  }
  {
    const int n = DD * 3 * DD;
    k_transpose_to_bf16<<<(n + 255) / 256, 256, 0, stream>>>(Wqkv, wqkvT, DD, 3 * DD);
  }
  {
    const int n = DD * DD;
    k_transpose_to_bf16<<<(n + 255) / 256, 256, 0, stream>>>(Wout, woutT, DD, DD);
  }
  k_gemm_qkv<<<dim3(3 * DD / 64, BB * TT / 64), 128, 0, stream>>>(
      xb, wqkvT, bqkv, qb, kbuf, vtb);
  k_attn<<<dim3(TT / 64, HH, BB), 128, 0, stream>>>(qb, kbuf, vtb, ctx);
  k_gemm_out<<<dim3(DD / 64, BB * TT / 64), 128, 0, stream>>>(ctx, woutT, bout, out);
}
